// CRW_18726057411442
// MI455X (gfx1250) — matrix operand order repaired and checked by reading, not compiled, not run on hardware
//
#include <hip/hip_runtime.h>
#include <math.h>

// MI455X / gfx1250, wave32. Heavy matmuls via v_wmma_f32_16x16x32_f16 with
// packed-f16 operands laid out so every fragment is 2x global_load_b128.
// Chain matmuls use 32x32 register blocking (4 accumulators / wave).

typedef __attribute__((ext_vector_type(16))) _Float16 v16h;
typedef __attribute__((ext_vector_type(8)))  _Float16 v8h;
typedef __attribute__((ext_vector_type(8)))  float    v8f;

#define B_    4
#define N_    784
#define C_    512
#define T_    8
#define D_    128
#define M_    785           // N+1
#define MP_   800           // M padded to 16*50 for WMMA tiling
#define NMAT_ 56            // 2 dirs * B * (T-1)
#define EPS_  1e-20f
#define LOSS_BLOCKS 13      // ceil(B_*M_/256)

// ---------------- WMMA fragment helpers (packed f16, unguarded) --------------
// A 16x32 (MxK), ISA 7.12.2: lane r (0-15) rows, half-lane kh in {0,8}:
// V0..3 = K{kh..kh+7}, V4..7 = K{kh+16..kh+23} -> two contiguous 16B chunks.
__device__ __forceinline__ v16h frag_a_h(const _Float16* __restrict__ src, int ld,
                                         int r0, int k0, int lane) {
  int r  = r0 + (lane & 15);
  int kh = (lane >> 4) * 8;
  const _Float16* p = src + (long)r * ld + k0 + kh;
  v8h lo = *(const v8h*)p;          // K = k0+kh .. +7
  v8h hi = *(const v8h*)(p + 16);   // K = k0+kh+16 .. +23
  return __builtin_shufflevector(lo, hi, 0, 1, 2, 3, 4, 5, 6, 7,
                                         8, 9, 10, 11, 12, 13, 14, 15);
}

// B 32x16 (KxN): lane c = column, half-lane K offset {0,8} (+16 for hi); K contiguous in
// storage (elem(k,c) = src[c*ld + k]) -> two contiguous 16B chunks.
__device__ __forceinline__ v16h frag_b_h(const _Float16* __restrict__ src, int ld,
                                         int k0, int c0, int lane) {
  int c  = c0 + (lane & 15);
  int kh = (lane >> 4) * 8;
  const _Float16* p = src + (long)c * ld + k0 + kh;
  v8h lo = *(const v8h*)p;          // K = k0+kh .. +7
  v8h hi = *(const v8h*)(p + 16);   // K = k0+kh+16 .. +23
  return __builtin_shufflevector(lo, hi, 0, 1, 2, 3, 4, 5, 6, 7,
                                         8, 9, 10, 11, 12, 13, 14, 15);
}

// C/D 16x16 f32: VGPR v, lanes 0-15 -> (r0+v, c), lanes 16-31 -> (r0+8+v, c)
__device__ __forceinline__ void store_f(float* __restrict__ dst, int ld,
                                        int r0, int c0, int lane, v8f d) {
  int c  = c0 + (lane & 15);
  int rb = r0 + 8 * (lane >> 4);
#pragma unroll
  for (int v = 0; v < 8; ++v) dst[(long)(rb + v) * ld + c] = d[v];
}

// f16 row-major store (scattered b16)
__device__ __forceinline__ void store_h(_Float16* __restrict__ dst, int ld,
                                        int r0, int c0, int lane, v8f d) {
  int c  = c0 + (lane & 15);
  int rb = r0 + 8 * (lane >> 4);
#pragma unroll
  for (int v = 0; v < 8; ++v) dst[(long)(rb + v) * ld + c] = (_Float16)d[v];
}

// f16 transposed store: dst[c*ld + r] -> 8 contiguous halves = one b128 store
__device__ __forceinline__ void store_hT(_Float16* __restrict__ dst, int ld,
                                         int r0, int c0, int lane, v8f d) {
  int c  = c0 + (lane & 15);
  int rb = r0 + 8 * (lane >> 4);
  v8h o;
#pragma unroll
  for (int v = 0; v < 8; ++v) o[v] = (_Float16)d[v];
  *(v8h*)(dst + (long)c * ld + rb) = o;
}

// ------------------- Stage 1: projection + L2 norm ---------------------------
// One block (128 thr, thread = d) per (b,t,m) row. m==784 is the dust column.
// Writes q (B,D,T,M) f32 to d_out and qh (b,t,m,d) packed f16 to workspace.
__global__ void CRW_proj(const float* __restrict__ feats, const float* __restrict__ dust,
                         const float* __restrict__ Ws, const float* __restrict__ bs,
                         const float* __restrict__ Wd, const float* __restrict__ bd,
                         float* __restrict__ q_out, _Float16* __restrict__ qh) {
  int idx = blockIdx.x;
  int m = idx % M_; idx /= M_;
  int t = idx % T_; idx /= T_;
  int b = idx;
  int d = threadIdx.x;

  float acc;
  if (m < N_) {
    const float* fr = feats + ((long)(b * N_ + m) * C_) * T_ + t;  // stride T_ over c
    acc = bs[d];
    for (int c = 0; c < C_; ++c) acc += fr[(long)c * T_] * Ws[(long)c * D_ + d];
  } else {
    const float* dr = dust + ((long)b * C_) * T_ + t;
    acc = bd[d];
    for (int c = 0; c < C_; ++c) acc += dr[(long)c * T_] * Wd[(long)c * D_ + d];
  }

  __shared__ float red[D_];
  red[d] = acc * acc;
  __syncthreads();
  for (int s = D_ / 2; s > 0; s >>= 1) {
    if (d < s) red[d] += red[d + s];
    __syncthreads();
  }
  float nrm = fmaxf(sqrtf(red[0]), 1e-12f);
  float val = acc / nrm;

  q_out[(((long)b * D_ + d) * T_ + t) * M_ + m] = val;
  qh[((long)(b * T_ + t) * M_ + m) * D_ + d]    = (_Float16)val;
}

// ------------------- Stage 2: affinity GEMMs (WMMA) --------------------------
// 56 GEMMs of 784x784, K=128: A0 = X_tA^T * X_tB. 784 = 49 exact tiles, and
// qh[n*D+d] serves as A-operand (row-major) AND B-operand (K-contiguous).
__global__ void CRW_aff_gemm(const _Float16* __restrict__ qh, float* __restrict__ A) {
  int lane = threadIdx.x & 31;
  int tile = blockIdx.x * 4 + (threadIdx.x >> 5);
  if (tile >= 49 * 49) return;
  int tn = (tile / 49) * 16;
  int tm = (tile % 49) * 16;

  int g = blockIdx.y;
  int dir = g / 28, rem = g % 28;
  int b = rem / 7, t = rem % 7;
  int tA = dir ? (T_ - 1 - t) : t;
  int tB = dir ? (T_ - 2 - t) : (t + 1);

  const _Float16* qA = qh + ((long)(b * T_ + tA) * M_) * D_;
  const _Float16* qB = qh + ((long)(b * T_ + tB) * M_) * D_;
  float* Am = A + (long)g * M_ * M_;

  v8f acc = {};
#pragma unroll
  for (int k0 = 0; k0 < D_; k0 += 32) {
    v16h af = frag_a_h(qA, D_, tn, k0, lane);
    v16h bf = frag_b_h(qB, D_, k0, tm, lane);
    acc = __builtin_amdgcn_wmma_f32_16x16x32_f16(false, af, false, bf,
                                                 (short)0, acc, false, false);
  }
  store_f(Am, M_, tn, tm, lane, acc);
}

// ------------------- Stage 3: entropy dust column ----------------------------
__global__ void CRW_rowstats(float* __restrict__ A) {
  long mat = blockIdx.y;
  int  n   = blockIdx.x;
  float* row = A + mat * (long)M_ * M_ + (long)n * M_;
  __shared__ float sd[256];

  float s = 0.f;
  for (int m = threadIdx.x; m < N_; m += 256) s += row[m];
  sd[threadIdx.x] = s; __syncthreads();
  for (int k = 128; k > 0; k >>= 1) { if (threadIdx.x < k) sd[threadIdx.x] += sd[threadIdx.x + k]; __syncthreads(); }
  float invS = 1.0f / sd[0];
  __syncthreads();

  float e = 0.f;
  for (int m = threadIdx.x; m < N_; m += 256) {
    float p = row[m] * invS;
    e += p * logf(fmaxf(p, EPS_));
  }
  sd[threadIdx.x] = e; __syncthreads();
  for (int k = 128; k > 0; k >>= 1) { if (threadIdx.x < k) sd[threadIdx.x] += sd[threadIdx.x + k]; __syncthreads(); }
  if (threadIdx.x == 0) row[N_] = (-sd[0]) / logf((float)N_);
}

// ------------------- Stage 4: dust row (negative column sums) ----------------
__global__ void CRW_colsum(float* __restrict__ A) {
  long mat = blockIdx.y;
  int  m   = blockIdx.x * 256 + threadIdx.x;
  if (m >= M_) return;
  float* Am = A + mat * (long)M_ * M_;
  if (m == N_) { Am[(long)N_ * M_ + N_] = 0.0f; return; }  // corner forced to 0
  float s = 0.f;
  for (int n = 0; n < N_; ++n) s += Am[(long)n * M_ + m];
  Am[(long)N_ * M_ + m] = -s;
}

// ------------------- Stage 5: row softmax with temperature (in place) --------
__global__ void CRW_softmax(float* __restrict__ A) {
  long mat = blockIdx.y;
  int  r   = blockIdx.x;
  float* row = A + mat * (long)M_ * M_ + (long)r * M_;
  const float invT = 1.0f / 0.07f;
  __shared__ float sd[256];

  float mx = -3.0e38f;
  for (int m = threadIdx.x; m < M_; m += 256) mx = fmaxf(mx, row[m] * invT);
  sd[threadIdx.x] = mx; __syncthreads();
  for (int k = 128; k > 0; k >>= 1) { if (threadIdx.x < k) sd[threadIdx.x] = fmaxf(sd[threadIdx.x], sd[threadIdx.x + k]); __syncthreads(); }
  mx = sd[0]; __syncthreads();

  float s = 0.f;
  for (int m = threadIdx.x; m < M_; m += 256) s += expf(row[m] * invT - mx);
  sd[threadIdx.x] = s; __syncthreads();
  for (int k = 128; k > 0; k >>= 1) { if (threadIdx.x < k) sd[threadIdx.x] += sd[threadIdx.x + k]; __syncthreads(); }
  float invZ = 1.0f / sd[0];

  for (int m = threadIdx.x; m < M_; m += 256) row[m] = expf(row[m] * invT - mx) * invZ;
}

// ------------------- Stage 6a: pack softmaxed mats to padded f16 -------------
// A12 (dir0) -> transposed f16 (B-operand of left chain)
// A21 (dir1) -> row-major  f16 (A-operand of right chain)
__global__ void CRW_pack(const float* __restrict__ A,
                         _Float16* __restrict__ A12T, _Float16* __restrict__ A21h) {
  int idx = blockIdx.x * 256 + threadIdx.x;            // < MP_*MP_
  int r = idx / MP_, c = idx % MP_;
  int g = blockIdx.y;                                  // 0..55
  const float* src = A + (long)g * M_ * M_;
  float v = (r < M_ && c < M_) ? src[(long)r * M_ + c] : 0.0f;
  if (g < 28) A12T[(long)g * MP_ * MP_ + (long)c * MP_ + r]        = (_Float16)v;
  else        A21h[(long)(g - 28) * MP_ * MP_ + (long)r * MP_ + c] = (_Float16)v;
}

// ------------------- Stage 6b: init chain state ------------------------------
// left0 = A12[:,0] row-major f16 ; right0 = A21[:,T-2] transposed f16
__global__ void CRW_initLR(const float* __restrict__ A,
                           _Float16* __restrict__ Lh, _Float16* __restrict__ RhT) {
  int idx = blockIdx.x * 256 + threadIdx.x;
  int i = idx / MP_, j = idx % MP_;
  int g = blockIdx.y;
  int b = g >> 1;
  if (!(g & 1)) {
    const float* src = A + (long)(b * 7) * M_ * M_;
    float v = (i < M_ && j < M_) ? src[(long)i * M_ + j] : 0.0f;
    Lh[(long)b * MP_ * MP_ + (long)i * MP_ + j] = (_Float16)v;
  } else {
    const float* src = A + (long)(28 + b * 7 + (T_ - 2)) * M_ * M_;
    float v = (i < M_ && j < M_) ? src[(long)i * M_ + j] : 0.0f;
    RhT[(long)b * MP_ * MP_ + (long)j * MP_ + i] = (_Float16)v;   // transposed
  }
}

// ------------------- Stage 7: chain matmul step (WMMA, 32x32 blocking) -------
// grid.y = 2*B (b, left/right); grid.x covers 25x25 32x32-blocks, 8 waves/blk.
//   left : Lnew = Lold @ A12[b,i]          (X=Lh row-major, Y=A12T)
//   right: Rnew = A21[b,T-2-i] @ Rold      (X=A21h,         Y=RhT)
// All operands 800-padded f16, zero in padding -> no guards anywhere.
// Each wave holds 2 A-frags + 2 B-frags -> 4 accumulators (4 wmma / k-step),
// halving operand traffic per output element vs 16x16 tiles.
__global__ void CRW_chain(const _Float16* __restrict__ A12T,
                          const _Float16* __restrict__ A21h,
                          const _Float16* __restrict__ LhC, const _Float16* __restrict__ RhC,
                          _Float16* __restrict__ LhN, _Float16* __restrict__ RhN,
                          float* __restrict__ Lf, float* __restrict__ Rf, int i) {
  int lane = threadIdx.x & 31;
  int tile = blockIdx.x * 8 + (threadIdx.x >> 5);
  if (tile >= 25 * 25) return;
  int r0 = (tile / 25) * 32;
  int c0 = (tile % 25) * 32;

  int g = blockIdx.y;
  int b = g >> 1;
  const long mpsq = (long)MP_ * MP_;

  const _Float16* X;
  const _Float16* Y;
  if (!(g & 1)) {
    X = LhC + (long)b * mpsq;
    Y = A12T + (long)(b * 7 + i) * mpsq;
  } else {
    X = A21h + (long)(b * 7 + (T_ - 2 - i)) * mpsq;
    Y = RhC + (long)b * mpsq;
  }

  v8f a00 = {}, a01 = {}, a10 = {}, a11 = {};
  for (int kk = 0; kk < MP_; kk += 32) {
    v16h x0 = frag_a_h(X, MP_, r0,      kk, lane);
    v16h x1 = frag_a_h(X, MP_, r0 + 16, kk, lane);
    v16h y0 = frag_b_h(Y, MP_, kk, c0,      lane);
    v16h y1 = frag_b_h(Y, MP_, kk, c0 + 16, lane);
    a00 = __builtin_amdgcn_wmma_f32_16x16x32_f16(false, x0, false, y0, (short)0, a00, false, false);
    a01 = __builtin_amdgcn_wmma_f32_16x16x32_f16(false, x0, false, y1, (short)0, a01, false, false);
    a10 = __builtin_amdgcn_wmma_f32_16x16x32_f16(false, x1, false, y0, (short)0, a10, false, false);
    a11 = __builtin_amdgcn_wmma_f32_16x16x32_f16(false, x1, false, y1, (short)0, a11, false, false);
  }

  if (!(g & 1)) {
    float*    Fo = Lf  + (long)b * mpsq;
    _Float16* Ho = LhN + (long)b * mpsq;
    store_f(Fo, MP_, r0,      c0,      lane, a00);
    store_f(Fo, MP_, r0,      c0 + 16, lane, a01);
    store_f(Fo, MP_, r0 + 16, c0,      lane, a10);
    store_f(Fo, MP_, r0 + 16, c0 + 16, lane, a11);
    store_h(Ho, MP_, r0,      c0,      lane, a00);
    store_h(Ho, MP_, r0,      c0 + 16, lane, a01);
    store_h(Ho, MP_, r0 + 16, c0,      lane, a10);
    store_h(Ho, MP_, r0 + 16, c0 + 16, lane, a11);
  } else {
    float*    Fo = Rf  + (long)b * mpsq;
    _Float16* Ho = RhN + (long)b * mpsq;
    store_f(Fo, MP_, r0,      c0,      lane, a00);
    store_f(Fo, MP_, r0,      c0 + 16, lane, a01);
    store_f(Fo, MP_, r0 + 16, c0,      lane, a10);
    store_f(Fo, MP_, r0 + 16, c0 + 16, lane, a11);
    store_hT(Ho, MP_, r0,      c0,      lane, a00);   // contiguous b128 each
    store_hT(Ho, MP_, r0,      c0 + 16, lane, a01);
    store_hT(Ho, MP_, r0 + 16, c0,      lane, a10);
    store_hT(Ho, MP_, r0 + 16, c0 + 16, lane, a11);
  }
}

// ------------------- Stage 8: loss pieces ------------------------------------
// v[b,j] = rowsum(right[b])[j]
__global__ void CRW_rowsumR(const float* __restrict__ R, float* __restrict__ v) {
  int idx = blockIdx.x * 256 + threadIdx.x;
  if (idx >= B_ * M_) return;
  int b = idx / M_, j = idx % M_;
  const float* row = R + (long)b * MP_ * MP_ + (long)j * MP_;
  float s = 0.f;
  for (int m = 0; m < M_; ++m) s += row[m];
  v[idx] = s;
}

// contrib = log(rowsum(AA)+M*EPS) - log(diag(AA)+EPS); fixed-order partials.
__global__ void CRW_loss_partial(const float* __restrict__ L, const float* __restrict__ R,
                                 const float* __restrict__ v, float* __restrict__ part,
                                 int step) {
  int idx = blockIdx.x * 256 + threadIdx.x;
  float contrib = 0.f;
  if (idx < B_ * M_) {
    int b = idx / M_, j = idx % M_;
    const float* lrow  = L + (long)b * MP_ * MP_ + (long)j * MP_;
    const float* rbase = R + (long)b * MP_ * MP_;
    const float* vb    = v + (long)b * M_;
    float rs = 0.f, dg = 0.f;
    for (int m = 0; m < M_; ++m) {
      float lm = lrow[m];
      rs += lm * vb[m];
      dg += lm * rbase[(long)m * MP_ + j];
    }
    contrib = logf(rs + (float)M_ * EPS_) - logf(dg + EPS_);
  }
  __shared__ float sd[256];
  sd[threadIdx.x] = contrib; __syncthreads();
  for (int k = 128; k > 0; k >>= 1) { if (threadIdx.x < k) sd[threadIdx.x] += sd[threadIdx.x + k]; __syncthreads(); }
  if (threadIdx.x == 0) part[step * LOSS_BLOCKS + blockIdx.x] = sd[0];
}

__global__ void CRW_loss_final(const float* __restrict__ part, float* __restrict__ loss) {
  if (threadIdx.x == 0) {
    float s = 0.f;
    for (int i = 0; i < 6 * LOSS_BLOCKS; ++i) s += part[i];
    *loss = s / (float)(B_ * M_ * 6);
  }
}

// -----------------------------------------------------------------------------
extern "C" void kernel_launch(void* const* d_in, const int* in_sizes, int n_in,
                              void* d_out, int out_size, void* d_ws, size_t ws_size,
                              hipStream_t stream) {
  (void)in_sizes; (void)n_in; (void)out_size; (void)ws_size;
  const float* feats = (const float*)d_in[0];
  const float* dustr = (const float*)d_in[1];
  const float* Ws    = (const float*)d_in[2];
  const float* bs    = (const float*)d_in[3];
  const float* Wd    = (const float*)d_in[4];
  const float* bd    = (const float*)d_in[5];

  float* out   = (float*)d_out;
  float* q_out = out;                                   // (B,D,T,M)
  float* loss  = out + (long)B_ * D_ * T_ * M_;         // scalar

  // Workspace carve (256B-aligned so v8h/b128 accesses are aligned)
  const long mpsq = (long)MP_ * MP_;
  char* wp = (char*)d_ws;
  auto carve = [&](size_t bytes) { char* r = wp; wp += (bytes + 255) & ~(size_t)255; return r; };
  _Float16* qh   = (_Float16*)carve((size_t)B_ * T_ * M_ * D_ * 2);   // packed f16 q
  float*    A    = (float*)   carve((size_t)NMAT_ * M_ * M_ * 4);     // raw/softmax mats
  _Float16* A12T = (_Float16*)carve((size_t)28 * mpsq * 2);           // dir0 transposed
  _Float16* A21h = (_Float16*)carve((size_t)28 * mpsq * 2);           // dir1 row-major
  _Float16* Lh0  = (_Float16*)carve((size_t)B_ * mpsq * 2);
  _Float16* Lh1  = (_Float16*)carve((size_t)B_ * mpsq * 2);
  _Float16* Rh0  = (_Float16*)carve((size_t)B_ * mpsq * 2);
  _Float16* Rh1  = (_Float16*)carve((size_t)B_ * mpsq * 2);
  float*    Lf   = (float*)   carve((size_t)B_ * mpsq * 4);           // newest left (f32)
  float*    Rf   = (float*)   carve((size_t)B_ * mpsq * 4);           // newest right (f32)
  float*    vv   = (float*)   carve((size_t)B_ * M_ * 4);
  float*    pt   = (float*)   carve((size_t)6 * LOSS_BLOCKS * 4);

  CRW_proj<<<B_ * T_ * M_, D_, 0, stream>>>(feats, dustr, Ws, bs, Wd, bd, q_out, qh);
  CRW_aff_gemm<<<dim3((49 * 49 + 3) / 4, NMAT_), 128, 0, stream>>>(qh, A);
  CRW_rowstats<<<dim3(N_, NMAT_), 256, 0, stream>>>(A);
  CRW_colsum<<<dim3(4, NMAT_), 256, 0, stream>>>(A);
  CRW_softmax<<<dim3(M_, NMAT_), 256, 0, stream>>>(A);
  CRW_pack<<<dim3((int)(mpsq / 256), NMAT_), 256, 0, stream>>>(A, A12T, A21h);
  CRW_initLR<<<dim3((int)(mpsq / 256), 2 * B_), 256, 0, stream>>>(A, Lh0, Rh0);

  _Float16* LhC = Lh0; _Float16* LhN = Lh1;
  _Float16* RhC = Rh0; _Float16* RhN = Rh1;
  for (int i = 1; i <= T_ - 2; ++i) {
    CRW_chain<<<dim3((25 * 25 + 7) / 8, 2 * B_), 256, 0, stream>>>(
        A12T, A21h, LhC, RhC, LhN, RhN, Lf, Rf, i);
    CRW_rowsumR<<<LOSS_BLOCKS, 256, 0, stream>>>(Rf, vv);
    CRW_loss_partial<<<LOSS_BLOCKS, 256, 0, stream>>>(Lf, Rf, vv, pt, i - 1);
    _Float16* tmp;
    tmp = LhC; LhC = LhN; LhN = tmp;
    tmp = RhC; RhC = RhN; RhN = tmp;
  }
  CRW_loss_final<<<1, 64, 0, stream>>>(pt, loss);
}
